// GptOssAttention_1176821039987
// MI455X (gfx1250) — matrix operand order repaired and checked by reading, not compiled, not run on hardware
//
#include <hip/hip_runtime.h>

typedef __attribute__((ext_vector_type(16))) _Float16 v16h;
typedef __attribute__((ext_vector_type(8)))  float    v8f;
typedef __attribute__((ext_vector_type(2)))  float    v2f;
typedef __attribute__((ext_vector_type(4)))  unsigned u32x4;
typedef __attribute__((ext_vector_type(8)))  int      i32x8;
typedef __attribute__((ext_vector_type(4)))  int      i32x4;

#define HID   2880   // hidden size (K of qkv gemm)
#define QKVD  5120   // D*(H+2*HKV)
#define NKV   8
#define GRP   8      // H / HKV
#define HD    64     // head dim
#define NTOK  128    // B*Q
#define NB    8
#define QLEN  16
#define SEQ   2048   // P*PAGE
#define PGSZ  16
#define PPB   128    // pages per sequence
#define AD    4096   // H*D
#define NEGM  (-1e9f)
#define KVPAD 68     // 64 + 4 dword pad per row (LDS bank spread, TDM pad)

// ------------------------------------------------ CDNA5 feature detection

#if __has_builtin(__builtin_amdgcn_global_load_async_to_lds_b128) && \
    __has_builtin(__builtin_amdgcn_s_wait_asynccnt)
#define HAVE_ASYNC 1
#endif
#if __has_builtin(__builtin_amdgcn_tensor_load_to_lds) && \
    __has_builtin(__builtin_amdgcn_s_wait_tensorcnt)
#define HAVE_TDM 1
#endif

#ifdef HAVE_TDM
#define NASY 2   // async ops in flight per wave per chunk (K only)
#else
#define NASY 4   // K + V via async path
#endif

typedef __attribute__((address_space(1))) void GLOBAL_AS;
typedef __attribute__((address_space(3))) void LDS_AS;
// Builtin wants pointer-to-int32x4 operands (from round-2 diagnostic).
typedef __attribute__((address_space(1))) i32x4* gasv4_t;
typedef __attribute__((address_space(3))) i32x4* lasv4_t;

#ifdef HAVE_ASYNC
#define ASYNC_COPY16(gsrc, ldst)                                               \
  __builtin_amdgcn_global_load_async_to_lds_b128(                              \
      (gasv4_t)(GLOBAL_AS*)(gsrc), (lasv4_t)(LDS_AS*)(ldst), 0, 0)
#else
#define ASYNC_COPY16(gsrc, ldst)                                               \
  do { *(float4*)(ldst) = *(const float4*)(gsrc); } while (0)
#endif

template <int N> __device__ __forceinline__ void wait_async() {
#ifdef HAVE_ASYNC
  __builtin_amdgcn_s_wait_asynccnt(N);
#endif
}
template <int N> __device__ __forceinline__ void wait_tensor() {
#ifdef HAVE_TDM
  __builtin_amdgcn_s_wait_tensorcnt(N);
#endif
}

#ifdef HAVE_TDM
// TDM: 2D tile, 16 rows x 64 dwords, row stride 512 dwords in memory,
// LDS padded +4 dwords per 64 (-> 68-dword rows, matches KVPAD).
__device__ __forceinline__ void tdm_load_tile_16x64(const float* gsrc,
                                                    void* ldst) {
  const unsigned long long ga = (unsigned long long)gsrc;
  u32x4 g0;
  g0.x = 1u;                                  // count=1, user mode, no gather
  g0.y = (unsigned)(unsigned long long)ldst;  // LDS byte address
  g0.z = (unsigned)ga;                        // global addr lo
  g0.w = (unsigned)(ga >> 32) | (2u << 30);   // global addr hi | type=2
  i32x8 g1;
  g1[0] = (2 << 16)                           // data_size = 4B
        | (1 << 20)                           // pad_enable
        | (5 << 22)                           // pad_interval: every 64 DW
        | (3 << 25);                          // pad_amount: 4 DW
  g1[1] = (64 << 16);                         // tensor_dim0 = 64 (bits 79:48 lo)
  g1[2] = (16 << 16);                         // tensor_dim1 = 16 (bits 111:80 lo)
  g1[3] = (64 << 16);                         // tile_dim0 = 64 (bits 127:112)
  g1[4] = 16;                                 // tile_dim1 = 16, tile_dim2 = 0
  g1[5] = 512;                                // tensor_dim0_stride = 512 elems
  g1[6] = 0;
  g1[7] = 0;
  const i32x4 z4 = {0, 0, 0, 0};
#if __clang_major__ >= 23
  const i32x8 z8 = {0, 0, 0, 0, 0, 0, 0, 0};
  __builtin_amdgcn_tensor_load_to_lds(g0, g1, z4, z4, z8, 0);
#else
  __builtin_amdgcn_tensor_load_to_lds(g0, g1, z4, z4, 0);
#endif
}
#endif

// ---------------------------------------------------------------- WMMA helpers

__device__ __forceinline__ v8f wmma16(v16h a, v16h b, v8f c) {
  return __builtin_amdgcn_wmma_f32_16x16x32_f16(false, a, false, b, (short)0, c,
                                                false, false);
}

__device__ __forceinline__ void cvt4(const float4 v, _Float16* d) {
  d[0] = (_Float16)v.x; d[1] = (_Float16)v.y;
  d[2] = (_Float16)v.z; d[3] = (_Float16)v.w;
}

// A fragment (16x32 f16): lane row = lane&15; lanes0-15 K[0..15],
// lanes16-31 K[16..31].  rowp = &src[row*ldk + k0], hl = lane>>4.
__device__ __forceinline__ v16h load_afrag(const float* rowp, int hl) {
  _Float16 t[16];
  const float4* p0 = reinterpret_cast<const float4*>(rowp + hl * 16);
  const float4* p1 = reinterpret_cast<const float4*>(rowp + hl * 16 + 8);
  cvt4(p0[0], t + 0); cvt4(p0[1], t + 4);
  cvt4(p1[0], t + 8); cvt4(p1[1], t + 12);
  v16h a;
  #pragma unroll
  for (int i = 0; i < 16; ++i) a[i] = t[i];
  return a;
}

// B fragment (32x16 f16): lane column = lane&15, K = (lane>>4)*16 + 0..15,
// p = 16 contiguous K-floats (global or LDS).
__device__ __forceinline__ v16h load_bfrag(const float* p) {
  _Float16 t[16];
  const float4* q = reinterpret_cast<const float4*>(p);
  cvt4(q[0], t + 0); cvt4(q[1], t + 4); cvt4(q[2], t + 8); cvt4(q[3], t + 12);
  v16h b;
  #pragma unroll
  for (int i = 0; i < 16; ++i) b[i] = t[i];
  return b;
}

// reductions within each 16-lane half of a wave32
__device__ __forceinline__ float hmax16(float x) {
  #pragma unroll
  for (int m = 1; m < 16; m <<= 1) x = fmaxf(x, __shfl_xor(x, m, 32));
  return x;
}
__device__ __forceinline__ float hsum16(float x) {
  #pragma unroll
  for (int m = 1; m < 16; m <<= 1) x += __shfl_xor(x, m, 32);
  return x;
}

// ------------------------------------------------- GEMM: out = A @ W^T + bias
__global__ void __launch_bounds__(256) gemm_f16wmma_bias(
    const float* __restrict__ A, const float* __restrict__ W,
    const float* __restrict__ bias, float* __restrict__ out,
    const int N, const int K) {
  const int wave = threadIdx.x >> 5;
  const int lane = threadIdx.x & 31;
  const int hl = lane >> 4, lr = lane & 15;
  const int tile = blockIdx.x * 8 + wave;
  const int mt = tile & 7;    // 8 M tiles (M = 128)
  const int nt = tile >> 3;   // N/16 tiles
  const float* arow = A + (mt * 16 + lr) * K;
  const float* wrow = W + (nt * 16 + lr) * K;
  v8f c = {};
  for (int k0 = 0; k0 < K; k0 += 32) {
    __builtin_prefetch(wrow + k0 + 256, 0, 1);   // global_prefetch_b8
    v16h af = load_afrag(arow + k0, hl);
    v16h bf = load_bfrag(wrow + k0 + hl * 16);
    c = wmma16(af, bf, c);
  }
  const int col = nt * 16 + lr;
  const float bv = bias[col];
  #pragma unroll
  for (int r = 0; r < 8; ++r)
    out[(mt * 16 + hl * 8 + r) * N + col] = c[r] + bv;
}

// ------------------------------------------------- paged flash attention
// grid = B*HKV; block = 8 waves (one GQA group each).  K/V 32-key tiles are
// staged once per block into LDS: K via async-to-LDS (per-wave slices),
// V via TDM tensor_load_to_lds (wave 0) when available; double-buffered so
// chunk i+1's DMA overlaps chunk i's WMMA + online softmax.

__device__ __forceinline__ void stage_kv(
    const float* __restrict__ kvc, const int* __restrict__ pages, const int h,
    const int s0, float kb[32][KVPAD], float vb[32][KVPAD], const int wave,
    const int hl, const int lr) {
  #pragma unroll
  for (int t = 0; t < 2; ++t) {
    const int key = wave * 4 + t * 2 + hl;            // 0..31 within chunk
    const int pg = pages[(s0 >> 4) + (key >> 4)];
    const float* ksrc =
        kvc + (((pg * 2 + 0) * PGSZ + (key & 15)) * NKV + h) * HD + lr * 4;
    ASYNC_COPY16(ksrc, &kb[key][lr * 4]);
#ifndef HAVE_TDM
    const float* vsrc =
        kvc + (((pg * 2 + 1) * PGSZ + (key & 15)) * NKV + h) * HD + lr * 4;
    ASYNC_COPY16(vsrc, &vb[key][lr * 4]);
#endif
  }
#ifdef HAVE_TDM
  if (wave == 0) {                                    // wave-uniform branch
    #pragma unroll
    for (int ph = 0; ph < 2; ++ph) {
      const int pg = __builtin_amdgcn_readfirstlane(pages[(s0 >> 4) + ph]);
      const float* gsrc = kvc + (((pg * 2 + 1) * PGSZ) * NKV + h) * HD;
      tdm_load_tile_16x64(gsrc, &vb[ph * 16][0]);
    }
  }
#endif
}

__global__ void __launch_bounds__(256) attn_paged_kernel(
    const float* __restrict__ qkv, const float* __restrict__ kvc,
    const float* __restrict__ mask, const float* __restrict__ sinks,
    const int* __restrict__ page_idx, const int* __restrict__ last_lens,
    float* __restrict__ out) {
  __shared__ float kbuf[2][32][KVPAD];      // K tiles, double buffered
  __shared__ float vbuf[2][32][KVPAD];      // V tiles, double buffered
  __shared__ _Float16 pbuf[8][16 * 32];     // per-wave P tile (C->A restage)

  const int wave = threadIdx.x >> 5;
  const int lane = threadIdx.x & 31;
  const int hl = lane >> 4, lr = lane & 15;
  const int b = blockIdx.x >> 3;
  const int h = blockIdx.x & 7;
  const int head = h * GRP + wave;
  const float scaling = 0.125f;             // 64^-0.5

  // Q A-fragments (K = d = 64 -> two 16x32 frags), pre-scaled.
  const float* qrow = qkv + (b * QLEN + lr) * QKVD + head * HD;
  v16h qa[2];
  #pragma unroll
  for (int f = 0; f < 2; ++f) {
    v16h a;
    const int k0 = f * 32;
    #pragma unroll
    for (int j = 0; j < 8; ++j) {
      a[j]     = (_Float16)(qrow[k0 + hl * 16 + j] * scaling);
      a[j + 8] = (_Float16)(qrow[k0 + hl * 16 + 8 + j] * scaling);
    }
    qa[f] = a;
  }

  float mrun[8], lrun[8];
  v8f oacc[4] = {{}, {}, {}, {}};
  #pragma unroll
  for (int r = 0; r < 8; ++r) { mrun[r] = 0.0f; lrun[r] = 0.0f; }

  const int valid_len = (PPB - 1) * PGSZ + last_lens[b];
  const int* pages = page_idx + b * PPB;
  _Float16* myp = pbuf[wave];

  // prologue: stage chunk 0 into buffer 0
  stage_kv(kvc, pages, h, 0, kbuf[0], vbuf[0], wave, hl, lr);

  for (int ci = 0; ci < SEQ / 32; ++ci) {
    const int s0 = ci * 32;
    const int bb = ci & 1;
    if (ci + 1 < SEQ / 32) {
      stage_kv(kvc, pages, h, s0 + 32, kbuf[bb ^ 1], vbuf[bb ^ 1],
               wave, hl, lr);
      wait_async<NASY>();     // chunk ci's K(/V) slices landed in LDS
      wait_tensor<2>();       // chunk ci's 2 TDM V tiles done (wave0 issued)
    } else {
      wait_async<0>();
      wait_tensor<0>();
    }
    __syncthreads();          // whole tile visible to all 8 waves

    float (*kb)[KVPAD] = kbuf[bb];
    float (*vb)[KVPAD] = vbuf[bb];

    // scores: Q[16x64] x K^T[64x32] as 2 tiles x 2 K-steps (B-frags from LDS)
    v16h kb00 = load_bfrag(&kb[lr][hl * 16]);
    v16h kb01 = load_bfrag(&kb[lr][32 + hl * 16]);
    v16h kb10 = load_bfrag(&kb[16 + lr][hl * 16]);
    v16h kb11 = load_bfrag(&kb[16 + lr][32 + hl * 16]);

    v8f sc0 = {}, sc1 = {};
    sc0 = wmma16(qa[0], kb00, sc0);
    sc0 = wmma16(qa[1], kb01, sc0);
    sc1 = wmma16(qa[0], kb10, sc1);
    sc1 = wmma16(qa[1], kb11, sc1);

    // mask + online softmax (C-layout row = hl*8 + r, col = key lr)
    #pragma unroll
    for (int r = 0; r < 8; ++r) {
      const int qi = r + hl * 8;
      const float* mrow = mask + (b * QLEN + qi) * SEQ;
      float mv0 = (s0 + lr < valid_len)      ? mrow[s0 + lr]      : NEGM;
      float mv1 = (s0 + 16 + lr < valid_len) ? mrow[s0 + 16 + lr] : NEGM;
      float x0 = sc0[r] + mv0;
      float x1 = sc1[r] + mv1;
      float tmax = hmax16(fmaxf(x0, x1));
      float mnew = fmaxf(mrun[r], tmax);          // running max clamped at 0
      float scale = __expf(mrun[r] - mnew);
      mrun[r] = mnew;
      float p0 = __expf(x0 - mnew);
      float p1 = __expf(x1 - mnew);
      lrun[r] = lrun[r] * scale + hsum16(p0 + p1);
      #pragma unroll
      for (int dt = 0; dt < 4; ++dt) oacc[dt][r] *= scale;
      myp[qi * 32 + lr]      = (_Float16)p0;
      myp[qi * 32 + 16 + lr] = (_Float16)p1;
    }

    // P A-frag from LDS (wave-private region: in-wave DS order suffices)
    v16h pa;
    #pragma unroll
    for (int j = 0; j < 8; ++j) {
      pa[j]     = myp[lr * 32 + hl * 16 + j];
      pa[j + 8] = myp[lr * 32 + hl * 16 + 8 + j];
    }

    // PV: P[16x32] x V[32x64] as 4 d-tiles (V B-frags from LDS)
    #pragma unroll
    for (int dt = 0; dt < 4; ++dt) {
      v16h vf;
      #pragma unroll
      for (int j = 0; j < 16; ++j)
        vf[j] = (_Float16)vb[hl * 16 + j][dt * 16 + lr];
      oacc[dt] = wmma16(pa, vf, oacc[dt]);
    }

    __syncthreads();          // all waves done reading buf[bb] before reuse
  }

  // sink term in the denominator, normalize, store
  const float sk = sinks[head];
  #pragma unroll
  for (int r = 0; r < 8; ++r) lrun[r] += __expf(sk - mrun[r]);
  #pragma unroll
  for (int r = 0; r < 8; ++r) {
    const int qi = r + hl * 8;
    const int tok = b * QLEN + qi;
    const float inv = 1.0f / lrun[r];
    #pragma unroll
    for (int dt = 0; dt < 4; ++dt)
      out[tok * AD + head * HD + dt * 16 + lr] = oacc[dt][r] * inv;
  }
}

// ---------------------------------------------------------------- probe kept:
// exact-f32 WMMA (16x16x4) is available with this signature (confirmed R1).
__global__ void probe_wmma_f32k4(const float* __restrict__ a,
                                 const float* __restrict__ b,
                                 float* __restrict__ d) {
  v2f av = *reinterpret_cast<const v2f*>(a + threadIdx.x * 2);
  v2f bv = *reinterpret_cast<const v2f*>(b + threadIdx.x * 2);
  v8f c = {};
  c = __builtin_amdgcn_wmma_f32_16x16x4_f32(false, av, false, bv, (short)0, c,
                                            false, false);
  *reinterpret_cast<v8f*>(d + threadIdx.x * 8) = c;
}

// ---------------------------------------------------------------- launch

extern "C" void kernel_launch(void* const* d_in, const int* in_sizes, int n_in,
                              void* d_out, int out_size, void* d_ws,
                              size_t ws_size, hipStream_t stream) {
  const float* hidden  = (const float*)d_in[0];
  const float* kvc     = (const float*)d_in[1];
  const float* amask   = (const float*)d_in[2];
  const float* w_qkv   = (const float*)d_in[3];
  const float* b_qkv   = (const float*)d_in[4];
  const float* w_o     = (const float*)d_in[5];
  const float* b_o     = (const float*)d_in[6];
  const float* sinks   = (const float*)d_in[7];
  const int* last_lens = (const int*)d_in[10];
  const int* page_idx  = (const int*)d_in[11];

  float* qkv_buf  = (float*)d_ws;                    // [128, 5120]
  float* attn_buf = qkv_buf + (size_t)NTOK * QKVD;   // [128, 4096]

  gemm_f16wmma_bias<<<QKVD / 16, 256, 0, stream>>>(hidden, w_qkv, b_qkv,
                                                   qkv_buf, QKVD, HID);
  attn_paged_kernel<<<NB * NKV, 256, 0, stream>>>(qkv_buf, kvc, amask, sinks,
                                                  page_idx, last_lens,
                                                  attn_buf);
  gemm_f16wmma_bias<<<HID / 16, 256, 0, stream>>>(attn_buf, w_o, b_o,
                                                  (float*)d_out, HID, AD);
}
